// GCNContactModel_30906584662040
// MI455X (gfx1250) — matrix operand order repaired and checked by reading, not compiled, not run on hardware
//
#include <hip/hip_runtime.h>
#include <hip/hip_bf16.h>

// ---------------- problem constants (match reference) ----------------
#define NN      131072      // nodes
#define NFEA    9           // atom categorical features
#define DIM     275         // dim_inner
#define DPAD    288         // padded dim (multiple of 32)
#define NEDGE   524288      // directed edges
#define NLAY    5           // conv layers
#define NGRAPH  2048
#define NPG     64
#define MT      (NN/16)     // 8192 M-tiles
#define KT      (DPAD/32)   // 9 K-tiles
#define NT      (DPAD/16)   // 18 N-tiles
#define NTW     6           // N-tiles per wave in GEMM (18 = 3 * 6)
#define BN_EPS  1e-5f

typedef __bf16 bf16x16 __attribute__((ext_vector_type(16)));
typedef float  f32x8   __attribute__((ext_vector_type(8)));

union FragU { unsigned short us[16]; uint4 q[2]; };

__device__ __forceinline__ unsigned short f2bf(float f) {
    union { float f; unsigned u; } v; v.f = f;
    unsigned r = v.u + 0x7FFFu + ((v.u >> 16) & 1u);   // round-to-nearest-even
    return (unsigned short)(r >> 16);
}

// ---------------- graph normalization ----------------
__global__ void k_deg_init(float* deg) {
    int i = blockIdx.x * blockDim.x + threadIdx.x;
    if (i < NN) deg[i] = 1.0f;                         // self loop
}
__global__ void k_deg_count(const int* __restrict__ ei, float* deg) {
    int e = blockIdx.x * blockDim.x + threadIdx.x;
    if (e < NEDGE) atomicAdd(&deg[ei[NEDGE + e]], 1.0f);   // col = target
}
__global__ void k_dinv(const float* __restrict__ deg, float* dinv) {
    int i = blockIdx.x * blockDim.x + threadIdx.x;
    if (i < NN) dinv[i] = rsqrtf(deg[i]);
}

// ---------------- atom encoder: h[n,d] = sum_f emb[f, x[n,f], d] ----------------
__global__ void k_encoder(const int* __restrict__ x, const float* __restrict__ emb,
                          float* __restrict__ h) {
    long long idx = (long long)blockIdx.x * blockDim.x + threadIdx.x;
    if (idx >= (long long)NN * DPAD) return;
    int d = (int)(idx % DPAD);
    int n = (int)(idx / DPAD);
    float s = 0.0f;
    if (d < DIM) {
        const int* xr = x + (long long)n * NFEA;
        #pragma unroll
        for (int f = 0; f < NFEA; ++f)
            s += emb[((long long)f * 128 + xr[f]) * DIM + d];
    }
    h[(long long)n * DPAD + d] = s;
}

// ---------------- fragment packing ----------------
// A-frag layout: element e of lane ->
//   k = (lane>=16?16:0) + e,  m = lane&15
// Buffer layout: [mtile][ktile][lane][16]
__global__ void k_pack_a(const float* __restrict__ src, unsigned short* __restrict__ frag) {
    int t = blockIdx.x * blockDim.x + threadIdx.x;           // MT*KT*32 threads
    if (t >= MT * KT * 32) return;
    int lane = t & 31;
    int kt   = (t >> 5) % KT;
    int mt   = t / (KT * 32);
    const float* srow = src + (long long)(mt * 16 + (lane & 15)) * DPAD + kt * 32;
    FragU f;
    #pragma unroll
    for (int e = 0; e < 16; ++e) {
        int kk = ((lane & 16) ? 16 : 0) + e;
        f.us[e] = f2bf(srow[kk]);
    }
    uint4* dst = (uint4*)(frag + (long long)t * 16);
    dst[0] = f.q[0]; dst[1] = f.q[1];
}

// B-frag layout: n = lane&15, k = (lane>=16?16:0) + e   (SWMMAC-style halves)
// Weights: B[k][n] = W[n][k] (xw = h @ W^T). 6 matrices: 5 conv + post.
// Buffer layout: [layer][ktile][ntile][lane][16]  (ntile-contiguous per K-step)
__global__ void k_pack_w(const float* __restrict__ convW, const float* __restrict__ postW,
                         unsigned short* __restrict__ frag) {
    int t = blockIdx.x * blockDim.x + threadIdx.x;           // 6*KT*NT*32 threads
    if (t >= 6 * KT * NT * 32) return;
    int lane = t & 31;
    int nt   = (t >> 5) % NT;
    int kt   = (t / (NT * 32)) % KT;
    int l    = t / (NT * 32 * KT);
    const float* W = (l < NLAY) ? (convW + (long long)l * DIM * DIM) : postW;
    int n = nt * 16 + (lane & 15);
    FragU f;
    #pragma unroll
    for (int e = 0; e < 16; ++e) {
        int k = kt * 32 + ((lane & 16) ? 16 : 0) + e;
        f.us[e] = (n < DIM && k < DIM) ? f2bf(W[(long long)n * DIM + k]) : (unsigned short)0;
    }
    uint4* dst = (uint4*)(frag + (long long)t * 16);
    dst[0] = f.q[0]; dst[1] = f.q[1];
}

// Decode B-frag: B[k][n] = hd[n][k]  (columns are node rows)
__global__ void k_pack_b(const float* __restrict__ src, unsigned short* __restrict__ frag) {
    int t = blockIdx.x * blockDim.x + threadIdx.x;           // MT*KT*32 threads
    if (t >= MT * KT * 32) return;
    int lane = t & 31;
    int kt   = (t >> 5) % KT;
    int nt   = t / (KT * 32);
    const float* srow = src + (long long)(nt * 16 + (lane & 15)) * DPAD + kt * 32;
    FragU f;
    #pragma unroll
    for (int e = 0; e < 16; ++e)
        f.us[e] = f2bf(srow[((lane & 16) ? 16 : 0) + e]);
    uint4* dst = (uint4*)(frag + (long long)t * 16);
    dst[0] = f.q[0]; dst[1] = f.q[1];
}

// ---------------- GEMM: out[m,n] = sum_k A[m,k]*B[k,n] (+bias[n]) ----------------
// One wave computes a 16 x (16*NTW) tile. All NTW B fragments of a K-step are
// staged into registers before the WMMA burst so the loads issue as one clause
// and the WMMAs retire against staggered loadcnt waits.
__global__ void k_gemm(const unsigned short* __restrict__ afrag,
                       const unsigned short* __restrict__ wfrag_layer,
                       const float* __restrict__ bias,
                       float* __restrict__ out) {
    int lane = threadIdx.x;
    int mt = blockIdx.x;
    int ntbase = blockIdx.y * NTW;
    const unsigned short* ap = afrag + ((long long)mt * KT) * 512 + lane * 16;
    const unsigned short* bp = wfrag_layer + ((long long)ntbase) * 512 + lane * 16;
    f32x8 acc[NTW] = {};
    #pragma unroll
    for (int kt = 0; kt < KT; ++kt) {
        bf16x16 a = *(const bf16x16*)(const void*)ap; ap += 512;
        bf16x16 b[NTW];
        #pragma unroll
        for (int j = 0; j < NTW; ++j)
            b[j] = *(const bf16x16*)(const void*)(bp + (long long)j * 512);
        #pragma unroll
        for (int j = 0; j < NTW; ++j)
            acc[j] = __builtin_amdgcn_wmma_f32_16x16x32_bf16(false, a, false, b[j],
                                                             (short)0, acc[j], false, false);
        bp += (long long)NT * 512;
    }
    int mbase = mt * 16 + ((lane & 16) ? 8 : 0);
    #pragma unroll
    for (int j = 0; j < NTW; ++j) {
        int n = (ntbase + j) * 16 + (lane & 15);
        float bv = (bias != nullptr && n < DIM) ? bias[n] : 0.0f;
        #pragma unroll
        for (int r = 0; r < 8; ++r)
            out[(long long)(mbase + r) * DPAD + n] = acc[j][r] + bv;
    }
}

// ---------------- message passing ----------------
// agg init: bias + self-loop term dinv[n]^2 * xw[n]
__global__ void k_agg_init(const float* __restrict__ xw, const float* __restrict__ dinv,
                           const float* __restrict__ b, float* __restrict__ agg) {
    long long idx = (long long)blockIdx.x * blockDim.x + threadIdx.x;
    if (idx >= (long long)NN * DPAD) return;
    int d = (int)(idx % DPAD);
    int n = (int)(idx / DPAD);
    float di = dinv[n];
    float v = di * di * xw[idx];
    if (d < DIM) v += b[d];
    agg[idx] = v;
}
// one wave per edge; lanes stride D (coalesced gather + coalesced atomics)
__global__ void k_scatter(const int* __restrict__ ei, const float* __restrict__ dinv,
                          const float* __restrict__ xw, float* __restrict__ agg) {
    int t = blockIdx.x * blockDim.x + threadIdx.x;
    int e = t >> 5, lane = t & 31;
    if (e >= NEDGE) return;
    int r = ei[e];            // source
    int c = ei[NEDGE + e];    // target
    float nrm = dinv[r] * dinv[c];
    const float* src = xw + (long long)r * DPAD;
    float* dst = agg + (long long)c * DPAD;
    for (int d = lane; d < DIM; d += 32)
        atomicAdd(&dst[d], nrm * src[d]);
}

// ---------------- batch norm ----------------
__global__ void k_bn_partial(const float* __restrict__ agg, float* __restrict__ part) {
    int col = threadIdx.x;                 // 288 threads
    long long base = (long long)blockIdx.x * (NN / 128);
    float s = 0.f, sq = 0.f;
    for (int r = 0; r < NN / 128; ++r) {
        float v = agg[(base + r) * DPAD + col];
        s += v; sq += v * v;
    }
    part[(long long)blockIdx.x * DPAD + col] = s;
    part[(long long)(128 + blockIdx.x) * DPAD + col] = sq;
}
__global__ void k_bn_final(const float* __restrict__ part, const float* __restrict__ gamma,
                           const float* __restrict__ beta, float* __restrict__ ss) {
    int col = threadIdx.x;
    float s = 0.f, sq = 0.f;
    for (int b = 0; b < 128; ++b) {
        s  += part[(long long)b * DPAD + col];
        sq += part[(long long)(128 + b) * DPAD + col];
    }
    float mu = s / (float)NN;
    float var = sq / (float)NN - mu * mu;
    float scale = 0.f, shift = 0.f;
    if (col < DIM) {
        scale = gamma[col] * rsqrtf(var + BN_EPS);
        shift = beta[col] - mu * scale;
    }
    ss[col] = scale; ss[DPAD + col] = shift;
}
__global__ void k_bn_apply(const float* __restrict__ agg, const float* __restrict__ ss,
                           float* __restrict__ h) {
    long long idx = (long long)blockIdx.x * blockDim.x + threadIdx.x;
    if (idx >= (long long)NN * DPAD) return;
    int d = (int)(idx % DPAD);
    float y = agg[idx] * ss[d] + ss[DPAD + d];
    h[idx] = y > 0.f ? y : 0.f;
}

// ---------------- decode: per-graph 64x64 Gram via WMMA ----------------
__global__ void k_decode(const unsigned short* __restrict__ afrag,
                         const unsigned short* __restrict__ bfrag,
                         float* __restrict__ out) {
    int g = blockIdx.x;
    int lane = threadIdx.x & 31, wave = threadIdx.x >> 5;
    #pragma unroll
    for (int s = 0; s < 2; ++s) {
        int t = wave * 2 + s;            // 16 tiles, 8 waves
        int mi = t >> 2, ni = t & 3;
        const unsigned short* ap = afrag + ((long long)(g * 4 + mi) * KT) * 512 + lane * 16;
        const unsigned short* bp = bfrag + ((long long)(g * 4 + ni) * KT) * 512 + lane * 16;
        f32x8 acc = {};
        #pragma unroll
        for (int kt = 0; kt < KT; ++kt) {
            bf16x16 a = *(const bf16x16*)(const void*)ap; ap += 512;
            bf16x16 b = *(const bf16x16*)(const void*)bp; bp += 512;
            acc = __builtin_amdgcn_wmma_f32_16x16x32_bf16(false, a, false, b,
                                                          (short)0, acc, false, false);
        }
        int m = mi * 16 + ((lane & 16) ? 8 : 0);
        int n = ni * 16 + (lane & 15);
        #pragma unroll
        for (int r = 0; r < 8; ++r)
            out[((long long)g * NPG + m + r) * NPG + n] = acc[r];
    }
}

// ---------------- host orchestration ----------------
extern "C" void kernel_launch(void* const* d_in, const int* in_sizes, int n_in,
                              void* d_out, int out_size, void* d_ws, size_t ws_size,
                              hipStream_t stream) {
    const int*   x_atoms = (const int*)d_in[0];
    const int*   ei      = (const int*)d_in[1];
    const float* emb     = (const float*)d_in[2];
    const float* convW   = (const float*)d_in[3];
    const float* convB   = (const float*)d_in[4];
    const float* gamma   = (const float*)d_in[5];
    const float* beta    = (const float*)d_in[6];
    const float* postW   = (const float*)d_in[7];
    const float* postB   = (const float*)d_in[8];
    float* out = (float*)d_out;

    size_t off = 0;
    auto alloc = [&](size_t bytes) -> void* {
        void* p = (char*)d_ws + off;
        off += (bytes + 255) & ~(size_t)255;
        return p;
    };
    unsigned short* afrag = (unsigned short*)alloc((size_t)NN * DPAD * 2);   // A fragments
    float*          xw    = (float*)alloc((size_t)NN * DPAD * 4);
    float*          agg   = (float*)alloc((size_t)NN * DPAD * 4);            // agg (BN reads it, writes h)
    float*          h     = (float*)alloc((size_t)NN * DPAD * 4);
    float*          deg   = (float*)alloc((size_t)NN * 4);
    float*          dinv  = (float*)alloc((size_t)NN * 4);
    unsigned short* wfrag = (unsigned short*)alloc((size_t)6 * NT * KT * 512 * 2);
    float*          part  = (float*)alloc((size_t)256 * DPAD * 4);
    float*          ss    = (float*)alloc((size_t)2 * DPAD * 4);
    (void)ws_size; (void)in_sizes; (void)n_in; (void)out_size;

    const long long ND = (long long)NN * DPAD;
    const int TB = 256;

    // graph normalization (fixed across layers)
    k_deg_init <<<(NN + TB - 1) / TB, TB, 0, stream>>>(deg);
    k_deg_count<<<(NEDGE + TB - 1) / TB, TB, 0, stream>>>(ei, deg);
    k_dinv     <<<(NN + TB - 1) / TB, TB, 0, stream>>>(deg, dinv);

    // encoder + weight packing
    k_encoder<<<(int)((ND + TB - 1) / TB), TB, 0, stream>>>(x_atoms, emb, h);
    k_pack_w <<<(6 * NT * KT * 32 + TB - 1) / TB, TB, 0, stream>>>(convW, postW, wfrag);

    dim3 ggrid(MT, NT / NTW);
    for (int l = 0; l < NLAY; ++l) {
        k_pack_a<<<(MT * KT * 32 + TB - 1) / TB, TB, 0, stream>>>(h, afrag);
        k_gemm  <<<ggrid, 32, 0, stream>>>(afrag, wfrag + (long long)l * NT * KT * 512,
                                           nullptr, xw);
        k_agg_init<<<(int)((ND + TB - 1) / TB), TB, 0, stream>>>(xw, dinv,
                                                                 convB + (long long)l * DIM, agg);
        k_scatter <<<(NEDGE * 32) / TB, TB, 0, stream>>>(ei, dinv, xw, agg);
        k_bn_partial<<<128, DPAD, 0, stream>>>(agg, part);
        k_bn_final  <<<1, DPAD, 0, stream>>>(part, gamma + (long long)l * DIM,
                                             beta + (long long)l * DIM, ss);
        k_bn_apply  <<<(int)((ND + TB - 1) / TB), TB, 0, stream>>>(agg, ss, h);
    }

    // post_mp linear (bias fused)
    k_pack_a<<<(MT * KT * 32 + TB - 1) / TB, TB, 0, stream>>>(h, afrag);
    k_gemm  <<<ggrid, 32, 0, stream>>>(afrag, wfrag + (long long)NLAY * NT * KT * 512,
                                       postB, xw);

    // decode: reuse agg as A-frags, afrag as B-frags
    unsigned short* afragD = (unsigned short*)agg;
    unsigned short* bfragD = afrag;
    k_pack_a<<<(MT * KT * 32 + TB - 1) / TB, TB, 0, stream>>>(xw, afragD);
    k_pack_b<<<(MT * KT * 32 + TB - 1) / TB, TB, 0, stream>>>(xw, bfragD);
    k_decode<<<NGRAPH, 256, 0, stream>>>(afragD, bfragD, out);
}
